// Actor_5789615915034
// MI455X (gfx1250) — matrix operand order repaired and checked by reading, not compiled, not run on hardware
//
#include <hip/hip_runtime.h>

// ---------------- problem constants ----------------
#define BB 16
#define NN 64
#define TTT 512
#define FF 4
#define CC 128
#define EE 4032              // NN*(NN-1)
#define EB (BB*EE)           // 64512 edge rows
#define LDX 136              // LDS row stride (halves) for [t][c] tiles; 272B -> 4-dword bank shift
#define BN_RSQ 0.9999950000374997f   // 1/sqrt(1+1e-5)

typedef _Float16 half_t;
typedef __attribute__((ext_vector_type(16))) _Float16 v16h;
typedef __attribute__((ext_vector_type(8)))  float    v8f;

// -------- gfx1250 async memory->LDS path (probe-gated, falls back to sync copy) --------
#if __has_builtin(__builtin_amdgcn_global_load_async_to_lds_b128) && \
    __has_builtin(__builtin_amdgcn_s_wait_asynccnt)
typedef int v4i_ __attribute__((ext_vector_type(4)));
typedef __attribute__((address_space(1))) v4i_* gas_p;
typedef __attribute__((address_space(3))) v4i_* las_p;
__device__ __forceinline__ void async_cp16(half_t* dst, const half_t* src) {
  __builtin_amdgcn_global_load_async_to_lds_b128((gas_p)(half_t*)src, (las_p)dst, 0, 0);
}
__device__ __forceinline__ void async_wait() { __builtin_amdgcn_s_wait_asynccnt(0); }
#else
__device__ __forceinline__ void async_cp16(half_t* dst, const half_t* src) {
  *(uint4*)dst = *(const uint4*)src;
}
__device__ __forceinline__ void async_wait() {}
#endif

// ---------------- WMMA helpers ----------------
__device__ __forceinline__ v8f wmma16(v16h a, v16h b, v8f c) {
  return __builtin_amdgcn_wmma_f32_16x16x32_f16(false, a, false, b, (short)0, c, false, false);
}

// A-fragment (16x32, 16-bit): lane<16 -> M=lane, K = kc+[0..7], kc+16+[0..7]
//                             lane>=16 -> M=lane-16, K = kc+8+[0..7], kc+24+[0..7]
__device__ __forceinline__ v16h frag_a(const half_t* p, int ld, int mb, int kc, int lane) {
  int m  = mb + (lane & 15);
  int ko = kc + ((lane >> 4) << 3);
  const half_t* q = p + (size_t)m * ld + ko;
  union { v16h v; uint4 u[2]; } w;
  w.u[0] = *(const uint4*)(q);
  w.u[1] = *(const uint4*)(q + 16);
  return w.v;
}

// B-fragment (32x16, 16-bit): lane<16 -> N=nb+lane, K=kc+[0..7], kc+16+[0..7]; lane>=16 -> N=nb+lane-16, K=kc+8+[0..7], kc+24+[0..7]
// source stored row-major [N][K] (activations: row=t, K=c ; weights: row=out, K=in)
__device__ __forceinline__ v16h frag_b(const half_t* p, int ld, int nb, int kc, int lane) {
  int n  = nb + (lane & 15);
  int ko = kc + ((lane >> 4) << 3);
  const half_t* q = p + (size_t)n * ld + ko;
  union { v16h v; uint4 u[2]; } w;
  w.u[0] = *(const uint4*)(q);
  w.u[1] = *(const uint4*)(q + 16);
  return w.v;
}

__device__ __forceinline__ float sigmoidf_(float x){ return 1.0f/(1.0f+__expf(-x)); }
__device__ __forceinline__ float eluf_(float x){ return x > 0.0f ? x : __expf(x) - 1.0f; }

// ---------------- weight converters ----------------
__global__ void k_cvt(const float* __restrict__ s, half_t* __restrict__ d, int n) {
  int i = blockIdx.x*256 + threadIdx.x;
  if (i < n) d[i] = (half_t)s[i];
}
// conv weight [co][ci][K] (f32) -> [k][co][ci] (f16)
__global__ void k_cvtw(const float* __restrict__ s, half_t* __restrict__ d, int Cout, int Cin, int K) {
  int i = blockIdx.x*256 + threadIdx.x;
  int tot = Cout*Cin*K;
  if (i < tot) {
    int k = i / (Cout*Cin);
    int rem = i - k*(Cout*Cin);
    int co = rem / Cin, ci = rem - co*Cin;
    d[i] = (half_t)s[(co*Cin + ci)*K + k];
  }
}

// ---------------- residual block 0 (cin=4, d=1/2) ----------------
__global__ __launch_bounds__(256) void k_block0(
    const float* __restrict__ xin, half_t* __restrict__ x1,
    const float* __restrict__ g1w, const float* __restrict__ g1b,
    const float* __restrict__ s1w, const float* __restrict__ s1b,
    const half_t* __restrict__ wg2, const half_t* __restrict__ ws2,  // [5][C][C] f16
    const float* __restrict__ g2b, const float* __restrict__ s2b,
    const float* __restrict__ skw, const float* __restrict__ skb,
    const float* __restrict__ bn1g, const float* __restrict__ bn1b,
    const float* __restrict__ bn2g, const float* __restrict__ bn2b)
{
  __shared__ __align__(16) float  xs0[76*4];         // t in [t0-12, t0+64)
  __shared__ __align__(16) half_t hs[72*LDX];        // h1, t in [t0-8, t0+64)
  __shared__ float wg1l[2560], ws1l[2560], skwl[512];
  const int tid = threadIdx.x;
  const int bn  = blockIdx.x;
  const int t0  = blockIdx.y * 64;

  for (int i = tid; i < 76*4; i += 256) {
    int row = i >> 2, f = i & 3;
    int tg = t0 - 12 + row;
    xs0[i] = (tg >= 0) ? xin[((size_t)bn*TTT + tg)*FF + f] : 0.0f;
  }
  for (int i = tid; i < 2560; i += 256) { wg1l[i] = g1w[i]; ws1l[i] = s1w[i]; }
  for (int i = tid; i < 512;  i += 256) skwl[i] = skw[i];
  __syncthreads();

  // stage 1 (VALU; K*cin = 20 MACs)
  for (int i = tid; i < 72*CC; i += 256) {
    int tl = i >> 7, c = i & 127;
    float g = g1b[c], s = s1b[c];
    #pragma unroll
    for (int k = 0; k < 5; ++k) {
      #pragma unroll
      for (int ci = 0; ci < 4; ++ci) {
        float xv = xs0[(tl + 4 - k)*4 + ci];
        g += wg1l[(c*4 + ci)*5 + k] * xv;
        s += ws1l[(c*4 + ci)*5 + k] * xv;
      }
    }
    float v = sigmoidf_(g) * tanhf(s);
    hs[tl*LDX + c] = (half_t)(v * (bn1g[c]*BN_RSQ) + bn1b[c]);
  }
  __syncthreads();

  // stage 2 (WMMA, d2 = 2)
  const int lane = tid & 31, wave = tid >> 5;
  const int mb = wave * 16;
  const int n = lane & 15, hi = lane >> 4;
  const v8f vz = {0.f,0.f,0.f,0.f,0.f,0.f,0.f,0.f};
  v8f accg[4], accs[4];
  #pragma unroll
  for (int t = 0; t < 4; ++t) { accg[t] = vz; accs[t] = vz; }
  for (int ch = 0; ch < 4; ++ch) {
    int kc = ch * 32;
    for (int tap = 0; tap < 5; ++tap) {
      v16h Ag = frag_a(wg2 + (size_t)tap*CC*CC, CC, mb, kc, lane);
      v16h As = frag_a(ws2 + (size_t)tap*CC*CC, CC, mb, kc, lane);
      #pragma unroll
      for (int t = 0; t < 4; ++t) {
        v16h Bf = frag_b(hs, LDX, 8 + t*16 - 2*tap, kc, lane);
        accg[t] = wmma16(Ag, Bf, accg[t]);
        accs[t] = wmma16(As, Bf, accs[t]);
      }
    }
  }
  #pragma unroll
  for (int t = 0; t < 4; ++t) {
    int xrow = (12 + t*16 + n)*4;
    float xv0 = xs0[xrow], xv1 = xs0[xrow+1], xv2 = xs0[xrow+2], xv3 = xs0[xrow+3];
    __align__(16) half_t o8[8];
    #pragma unroll
    for (int r = 0; r < 8; ++r) {
      int c = mb + 8*hi + r;
      float g = accg[t][r] + g2b[c];
      float s = accs[t][r] + s2b[c];
      float v = sigmoidf_(g) * tanhf(s);
      v = v * (bn2g[c]*BN_RSQ) + bn2b[c];
      v += skb[c] + skwl[c*4]*xv0 + skwl[c*4+1]*xv1 + skwl[c*4+2]*xv2 + skwl[c*4+3]*xv3;
      o8[r] = (half_t)v;
    }
    *(uint4*)(x1 + ((size_t)bn*TTT + t0 + t*16 + n)*CC + mb + 8*hi) = *(const uint4*)o8;
  }
}

// ---------------- residual block 1 (cin=128, d=4/8) ----------------
__global__ __launch_bounds__(256) void k_block1(
    const half_t* __restrict__ x1, half_t* __restrict__ x2,
    const half_t* __restrict__ wg1, const half_t* __restrict__ ws1,
    const half_t* __restrict__ wg2, const half_t* __restrict__ ws2,
    const half_t* __restrict__ wsk,
    const float* __restrict__ g1b, const float* __restrict__ s1b,
    const float* __restrict__ g2b, const float* __restrict__ s2b,
    const float* __restrict__ skb,
    const float* __restrict__ bn1g, const float* __restrict__ bn1b,
    const float* __restrict__ bn2g, const float* __restrict__ bn2b)
{
  __shared__ __align__(16) half_t xs[112*LDX];   // x1, t in [t0-48, t0+64)
  __shared__ __align__(16) half_t hs[96*LDX];    // h1, t in [t0-32, t0+64)
  const int tid = threadIdx.x;
  const int bn  = blockIdx.x;
  const int t0  = blockIdx.y * 64;

  // stage-in x1 tile: async memory->LDS on gfx1250 (ASYNCcnt), zeros for causal pad
  for (int i = tid; i < 112*16; i += 256) {
    int row = i >> 4, col = (i & 15) * 8;
    int tg = t0 - 48 + row;
    half_t* dst = xs + row*LDX + col;
    if (tg >= 0) {
      async_cp16(dst, x1 + ((size_t)bn*TTT + tg)*CC + col);
    } else {
      uint4 z = {0u,0u,0u,0u};
      *(uint4*)dst = z;
    }
  }
  async_wait();
  __syncthreads();

  const int lane = tid & 31, wave = tid >> 5;
  const int mb = wave * 16;
  const int n = lane & 15, hi = lane >> 4;
  const v8f vz = {0.f,0.f,0.f,0.f,0.f,0.f,0.f,0.f};

  { // stage 1: d=4, 6 output t-tiles
    v8f ag[6], as_[6];
    #pragma unroll
    for (int t = 0; t < 6; ++t) { ag[t] = vz; as_[t] = vz; }
    for (int ch = 0; ch < 4; ++ch) {
      int kc = ch*32;
      for (int tap = 0; tap < 5; ++tap) {
        v16h Ag = frag_a(wg1 + (size_t)tap*CC*CC, CC, mb, kc, lane);
        v16h As = frag_a(ws1 + (size_t)tap*CC*CC, CC, mb, kc, lane);
        #pragma unroll
        for (int t = 0; t < 6; ++t) {
          v16h Bf = frag_b(xs, LDX, 16 + t*16 - 4*tap, kc, lane);
          ag[t]  = wmma16(Ag, Bf, ag[t]);
          as_[t] = wmma16(As, Bf, as_[t]);
        }
      }
    }
    #pragma unroll
    for (int t = 0; t < 6; ++t) {
      __align__(16) half_t o8[8];
      #pragma unroll
      for (int r = 0; r < 8; ++r) {
        int c = mb + 8*hi + r;
        float g = ag[t][r]  + g1b[c];
        float s = as_[t][r] + s1b[c];
        float v = sigmoidf_(g) * tanhf(s);
        o8[r] = (half_t)(v * (bn1g[c]*BN_RSQ) + bn1b[c]);
      }
      *(uint4*)(hs + (t*16 + n)*LDX + mb + 8*hi) = *(const uint4*)o8;
    }
  }
  __syncthreads();

  { // stage 2: d2=8 + fused 1x1 skip GEMM
    v8f ag[4], as_[4], ak[4];
    #pragma unroll
    for (int t = 0; t < 4; ++t) { ag[t] = vz; as_[t] = vz; ak[t] = vz; }
    for (int ch = 0; ch < 4; ++ch) {
      int kc = ch*32;
      v16h Ak = frag_a(wsk, CC, mb, kc, lane);
      #pragma unroll
      for (int t = 0; t < 4; ++t) {
        v16h Bx = frag_b(xs, LDX, 48 + t*16, kc, lane);
        ak[t] = wmma16(Ak, Bx, ak[t]);
      }
      for (int tap = 0; tap < 5; ++tap) {
        v16h Ag = frag_a(wg2 + (size_t)tap*CC*CC, CC, mb, kc, lane);
        v16h As = frag_a(ws2 + (size_t)tap*CC*CC, CC, mb, kc, lane);
        #pragma unroll
        for (int t = 0; t < 4; ++t) {
          v16h Bf = frag_b(hs, LDX, 32 + t*16 - 8*tap, kc, lane);
          ag[t]  = wmma16(Ag, Bf, ag[t]);
          as_[t] = wmma16(As, Bf, as_[t]);
        }
      }
    }
    #pragma unroll
    for (int t = 0; t < 4; ++t) {
      __align__(16) half_t o8[8];
      #pragma unroll
      for (int r = 0; r < 8; ++r) {
        int c = mb + 8*hi + r;
        float g = ag[t][r]  + g2b[c];
        float s = as_[t][r] + s2b[c];
        float v = sigmoidf_(g) * tanhf(s);
        v = v * (bn2g[c]*BN_RSQ) + bn2b[c];
        v += ak[t][r] + skb[c];
        o8[r] = (half_t)v;
      }
      *(uint4*)(x2 + ((size_t)bn*TTT + t0 + t*16 + n)*CC + mb + 8*hi) = *(const uint4*)o8;
    }
  }
}

// ---------------- pred/att/softmax/mean -> nodes (algebraically reduced) ----------------
__global__ __launch_bounds__(256) void k_nodes(
    const half_t* __restrict__ x2,
    const float* __restrict__ pw, const float* __restrict__ pb,
    const float* __restrict__ aw, const float* __restrict__ ab,
    float* __restrict__ nodes_f, half_t* __restrict__ nodes_h)
{
  __shared__ float att[TTT];
  __shared__ float wa[CC];
  __shared__ float yb[CC];
  __shared__ float red[256];
  const int tid = threadIdx.x;
  const size_t bn = blockIdx.x;
  if (tid < CC) wa[tid] = aw[tid];
  __syncthreads();
  float ab0 = ab[0];
  for (int t = tid; t < TTT; t += 256) {
    const half_t* xp = x2 + (bn*TTT + t)*CC;
    float a = ab0;
    for (int c = 0; c < CC; ++c) a += wa[c]*(float)xp[c];
    att[t] = a;
  }
  __syncthreads();
  float m = -3.4e38f;
  for (int t = tid; t < TTT; t += 256) m = fmaxf(m, att[t]);
  red[tid] = m; __syncthreads();
  for (int s = 128; s > 0; s >>= 1) { if (tid < s) red[tid] = fmaxf(red[tid], red[tid+s]); __syncthreads(); }
  m = red[0]; __syncthreads();
  float ssum = 0.0f;
  for (int t = tid; t < TTT; t += 256) { float e = __expf(att[t]-m); att[t] = e; ssum += e; }
  red[tid] = ssum; __syncthreads();
  for (int s = 128; s > 0; s >>= 1) { if (tid < s) red[tid] += red[tid+s]; __syncthreads(); }
  float inv = 1.0f / red[0];
  __syncthreads();
  for (int t = tid; t < TTT; t += 256) att[t] *= inv;
  __syncthreads();
  if (tid < CC) {               // y = x2 @ softmax  (coalesced over c)
    float acc = 0.0f;
    for (int t = 0; t < TTT; ++t) acc += (float)x2[(bn*TTT + t)*CC + tid] * att[t];
    yb[tid] = acc;
  }
  __syncthreads();
  {                             // nodes = (Wp @ y + pb) / T
    int d = tid;
    float acc = pb[d];
    for (int c = 0; c < CC; ++c) acc += pw[d*CC + c]*yb[c];
    acc *= (1.0f/512.0f);
    nodes_f[bn*256 + d] = acc;
    nodes_h[bn*256 + d] = (half_t)acc;
  }
}

// ---------------- edge input: recv*send ----------------
__global__ __launch_bounds__(256) void k_edgein(const float* __restrict__ nf, half_t* __restrict__ rs)
{
  size_t row = blockIdx.x;
  int b = (int)(row / EE), e = (int)(row % EE);
  int r = e / (NN-1), j = e % (NN-1);
  int s = j + (j >= r ? 1 : 0);
  int d = threadIdx.x;
  float v = nf[((size_t)b*NN + r)*256 + d] * nf[((size_t)b*NN + s)*256 + d];
  rs[row*256 + d] = (half_t)v;
}

// ---------------- generic fused two-layer ELU MLP + BN scale (WMMA) ----------------
template<int NIN>
__global__ __launch_bounds__(256) void k_mlp(
    const half_t* __restrict__ X,  const half_t* __restrict__ W1, const float* __restrict__ b1,
    const half_t* __restrict__ W2, const float* __restrict__ b2,
    const float* __restrict__ gg,  const float* __restrict__ bb,
    half_t* __restrict__ Y)
{
  constexpr int SX = NIN + 8;
  __shared__ __align__(16) half_t Xs[32*SX];
  __shared__ __align__(16) half_t Hs[32*264];
  const int tid = threadIdx.x, lane = tid & 31, wave = tid >> 5;
  const size_t row0 = (size_t)blockIdx.x * 32;
  // stage-in 32 input rows: async memory->LDS on gfx1250
  for (int i = tid; i < 32*(NIN/8); i += 256) {
    int r = i / (NIN/8), c = (i % (NIN/8)) * 8;
    async_cp16(Xs + r*SX + c, X + (row0 + r)*NIN + c);
  }
  async_wait();
  __syncthreads();
  const int n_ = lane & 15, hi = lane >> 4;
  const v8f vz = {0.f,0.f,0.f,0.f,0.f,0.f,0.f,0.f};
  for (int task = wave; task < 32; task += 8) {          // GEMM1 + ELU
    int mb = (task >> 4) * 16, nb = (task & 15) * 16;
    v8f acc = vz;
    for (int ch = 0; ch < NIN/32; ++ch) {
      v16h A  = frag_a(Xs, SX, mb, ch*32, lane);
      v16h Bf = frag_b(W1, NIN, nb, ch*32, lane);
      acc = wmma16(A, Bf, acc);
    }
    int nn = nb + n_;
    float bias = b1[nn];
    #pragma unroll
    for (int r = 0; r < 8; ++r)
      Hs[(mb + 8*hi + r)*264 + nn] = (half_t)eluf_(acc[r] + bias);
  }
  __syncthreads();
  for (int task = wave; task < 32; task += 8) {          // GEMM2 + ELU + BN scale
    int mb = (task >> 4) * 16, nb = (task & 15) * 16;
    v8f acc = vz;
    for (int ch = 0; ch < 8; ++ch) {
      v16h A  = frag_a(Hs, 264, mb, ch*32, lane);
      v16h Bf = frag_b(W2, 256, nb, ch*32, lane);
      acc = wmma16(A, Bf, acc);
    }
    int nn = nb + n_;
    float bias = b2[nn], sc = gg[nn]*BN_RSQ, sh = bb[nn];
    #pragma unroll
    for (int r = 0; r < 8; ++r) {
      float v = eluf_(acc[r] + bias);
      Xs[(mb + 8*hi + r)*SX + nn] = (half_t)(v*sc + sh);
    }
  }
  __syncthreads();
  for (int i = tid; i < 32*32; i += 256) {
    int r = i >> 5, c = (i & 31) * 8;
    *(uint4*)(Y + (row0 + r)*256 + c) = *(const uint4*)(Xs + r*SX + c);
  }
}

// ---------------- edge2node aggregation + concat [nodes|agg] ----------------
__global__ __launch_bounds__(256) void k_cat1(const half_t* __restrict__ edges,
                                              const half_t* __restrict__ nh,
                                              half_t* __restrict__ cat1)
{
  int ba = blockIdx.x;               // b*NN + a
  int b = ba / NN, a = ba % NN;
  int d = threadIdx.x;
  float acc = 0.0f;
  for (int j = 0; j < NN-1; ++j)
    acc += (float)edges[(((size_t)b*EE) + (size_t)a*(NN-1) + j)*256 + d];
  acc *= (1.0f/NN);
  cat1[(size_t)ba*512 + d]       = nh[(size_t)ba*256 + d];
  cat1[(size_t)ba*512 + 256 + d] = (half_t)acc;
}

// ---------------- concat [edges | recv2*send2] ----------------
__global__ __launch_bounds__(256) void k_cat2(const half_t* __restrict__ edges,
                                              const half_t* __restrict__ n2,
                                              half_t* __restrict__ cat2)
{
  size_t row = blockIdx.x;
  int b = (int)(row / EE), e = (int)(row % EE);
  int r = e / (NN-1), j = e % (NN-1);
  int s = j + (j >= r ? 1 : 0);
  int d = threadIdx.x;
  cat2[row*512 + d] = edges[row*256 + d];
  float v = (float)n2[((size_t)b*NN + r)*256 + d] * (float)n2[((size_t)b*NN + s)*256 + d];
  cat2[row*512 + 256 + d] = (half_t)v;
}

// ---------------- final 256->1 projection ----------------
__global__ __launch_bounds__(256) void k_out(const half_t* __restrict__ e2,
                                             const float* __restrict__ ow, const float* __restrict__ ob,
                                             float* __restrict__ out)
{
  __shared__ float w[256];
  int tid = threadIdx.x;
  w[tid] = ow[tid];
  __syncthreads();
  size_t row = (size_t)blockIdx.x*256 + tid;
  float acc = ob[0];
  const half_t* p = e2 + row*256;
  for (int d = 0; d < 256; ++d) acc += (float)p[d]*w[d];
  out[row] = acc;
}

// ================= launch =================
extern "C" void kernel_launch(void* const* d_in, const int* in_sizes, int n_in,
                              void* d_out, int out_size, void* d_ws, size_t ws_size,
                              hipStream_t stream)
{
  (void)in_sizes; (void)n_in; (void)out_size; (void)ws_size;
  // d_in order: inputs, rel_rec, rel_send, then params leaves in dict order.
  const float* inp = (const float*)d_in[0];
  // block0: 3..16 ; block1: 17..30
  const float *b0_g1w=(const float*)d_in[3],  *b0_g1b=(const float*)d_in[4];
  const float *b0_s1w=(const float*)d_in[5],  *b0_s1b=(const float*)d_in[6];
  const float *b0_g2w=(const float*)d_in[7],  *b0_g2b=(const float*)d_in[8];
  const float *b0_s2w=(const float*)d_in[9],  *b0_s2b=(const float*)d_in[10];
  const float *b0_skw=(const float*)d_in[11], *b0_skb=(const float*)d_in[12];
  const float *b0_bn1g=(const float*)d_in[13],*b0_bn1b=(const float*)d_in[14];
  const float *b0_bn2g=(const float*)d_in[15],*b0_bn2b=(const float*)d_in[16];
  const float *b1_g1w=(const float*)d_in[17], *b1_g1b=(const float*)d_in[18];
  const float *b1_s1w=(const float*)d_in[19], *b1_s1b=(const float*)d_in[20];
  const float *b1_g2w=(const float*)d_in[21], *b1_g2b=(const float*)d_in[22];
  const float *b1_s2w=(const float*)d_in[23], *b1_s2b=(const float*)d_in[24];
  const float *b1_skw=(const float*)d_in[25], *b1_skb=(const float*)d_in[26];
  const float *b1_bn1g=(const float*)d_in[27],*b1_bn1b=(const float*)d_in[28];
  const float *b1_bn2g=(const float*)d_in[29],*b1_bn2b=(const float*)d_in[30];
  const float *predw=(const float*)d_in[31], *predb=(const float*)d_in[32];
  const float *attw =(const float*)d_in[33], *attb =(const float*)d_in[34];
  const float *e1w1=(const float*)d_in[35], *e1b1=(const float*)d_in[36];
  const float *e1w2=(const float*)d_in[37], *e1b2=(const float*)d_in[38];
  const float *e1g =(const float*)d_in[39], *e1be=(const float*)d_in[40];
  const float *n1w1=(const float*)d_in[41], *n1b1=(const float*)d_in[42];
  const float *n1w2=(const float*)d_in[43], *n1b2=(const float*)d_in[44];
  const float *n1g =(const float*)d_in[45], *n1be=(const float*)d_in[46];
  const float *e2w1=(const float*)d_in[47], *e2b1=(const float*)d_in[48];
  const float *e2w2=(const float*)d_in[49], *e2b2=(const float*)d_in[50];
  const float *e2g =(const float*)d_in[51], *e2be=(const float*)d_in[52];
  const float *outw=(const float*)d_in[53], *outb=(const float*)d_in[54];

  char* base = (char*)d_ws;
  size_t off = 0;
  auto alloc = [&](size_t bytes)->char* {
    char* p = base + off; off = (off + bytes + 255) & ~(size_t)255; return p;
  };
  half_t* wb0g2 = (half_t*)alloc((size_t)5*CC*CC*2);
  half_t* wb0s2 = (half_t*)alloc((size_t)5*CC*CC*2);
  half_t* wb1g1 = (half_t*)alloc((size_t)5*CC*CC*2);
  half_t* wb1s1 = (half_t*)alloc((size_t)5*CC*CC*2);
  half_t* wb1g2 = (half_t*)alloc((size_t)5*CC*CC*2);
  half_t* wb1s2 = (half_t*)alloc((size_t)5*CC*CC*2);
  half_t* wb1sk = (half_t*)alloc((size_t)CC*CC*2);
  half_t* we1w1 = (half_t*)alloc((size_t)256*256*2);
  half_t* we1w2 = (half_t*)alloc((size_t)256*256*2);
  half_t* wn1w1 = (half_t*)alloc((size_t)256*512*2);
  half_t* wn1w2 = (half_t*)alloc((size_t)256*256*2);
  half_t* we2w1 = (half_t*)alloc((size_t)256*512*2);
  half_t* we2w2 = (half_t*)alloc((size_t)256*256*2);
  half_t* x1    = (half_t*)alloc((size_t)BB*NN*TTT*CC*2);
  half_t* x2    = (half_t*)alloc((size_t)BB*NN*TTT*CC*2);
  float*  nodesf= (float*) alloc((size_t)BB*NN*256*4);
  half_t* nodesh= (half_t*)alloc((size_t)BB*NN*256*2);
  half_t* rs    = (half_t*)alloc((size_t)EB*256*2);
  half_t* edgesh= (half_t*)alloc((size_t)EB*256*2);
  half_t* cat1  = (half_t*)alloc((size_t)BB*NN*512*2);
  half_t* n2h   = (half_t*)alloc((size_t)BB*NN*256*2);
  half_t* cat2  = (half_t*)alloc((size_t)EB*512*2);
  half_t* e2h   = (half_t*)alloc((size_t)EB*256*2);

  const int GW = (5*CC*CC + 255)/256;
  k_cvtw<<<GW,256,0,stream>>>(b0_g2w, wb0g2, CC, CC, 5);
  k_cvtw<<<GW,256,0,stream>>>(b0_s2w, wb0s2, CC, CC, 5);
  k_cvtw<<<GW,256,0,stream>>>(b1_g1w, wb1g1, CC, CC, 5);
  k_cvtw<<<GW,256,0,stream>>>(b1_s1w, wb1s1, CC, CC, 5);
  k_cvtw<<<GW,256,0,stream>>>(b1_g2w, wb1g2, CC, CC, 5);
  k_cvtw<<<GW,256,0,stream>>>(b1_s2w, wb1s2, CC, CC, 5);
  k_cvt<<<(CC*CC+255)/256,256,0,stream>>>(b1_skw, wb1sk, CC*CC);
  k_cvt<<<(65536+255)/256,256,0,stream>>>(e1w1, we1w1, 65536);
  k_cvt<<<(65536+255)/256,256,0,stream>>>(e1w2, we1w2, 65536);
  k_cvt<<<(131072+255)/256,256,0,stream>>>(n1w1, wn1w1, 131072);
  k_cvt<<<(65536+255)/256,256,0,stream>>>(n1w2, wn1w2, 65536);
  k_cvt<<<(131072+255)/256,256,0,stream>>>(e2w1, we2w1, 131072);
  k_cvt<<<(65536+255)/256,256,0,stream>>>(e2w2, we2w2, 65536);

  dim3 gblk(BB*NN, TTT/64);
  k_block0<<<gblk,256,0,stream>>>(inp, x1, b0_g1w,b0_g1b,b0_s1w,b0_s1b,
                                  wb0g2, wb0s2, b0_g2b,b0_s2b, b0_skw,b0_skb,
                                  b0_bn1g,b0_bn1b,b0_bn2g,b0_bn2b);
  k_block1<<<gblk,256,0,stream>>>(x1, x2, wb1g1,wb1s1,wb1g2,wb1s2,wb1sk,
                                  b1_g1b,b1_s1b,b1_g2b,b1_s2b,b1_skb,
                                  b1_bn1g,b1_bn1b,b1_bn2g,b1_bn2b);
  k_nodes<<<BB*NN,256,0,stream>>>(x2, predw,predb,attw,attb, nodesf, nodesh);
  k_edgein<<<EB,256,0,stream>>>(nodesf, rs);
  k_mlp<256><<<EB/32,256,0,stream>>>(rs, we1w1, e1b1, we1w2, e1b2, e1g, e1be, edgesh);
  k_cat1<<<BB*NN,256,0,stream>>>(edgesh, nodesh, cat1);
  k_mlp<512><<<(BB*NN)/32,256,0,stream>>>(cat1, wn1w1, n1b1, wn1w2, n1b2, n1g, n1be, n2h);
  k_cat2<<<EB,256,0,stream>>>(edgesh, n2h, cat2);
  k_mlp<512><<<EB/32,256,0,stream>>>(cat2, we2w1, e2b1, we2w2, e2b2, e2g, e2be, e2h);
  k_out<<<EB/256,256,0,stream>>>(e2h, outw, outb, (float*)d_out);
}
